// GraphAttentionNetwork_41721312313527
// MI455X (gfx1250) — compile-verified, matrix operand order checked by reading, not run on hardware
//
#include <hip/hip_runtime.h>
#include <cstdint>

typedef __attribute__((ext_vector_type(16))) __bf16 v16bf;
typedef __attribute__((ext_vector_type(8)))  float  v8f;
typedef __attribute__((ext_vector_type(4)))  float  f32x4;
typedef __attribute__((ext_vector_type(4)))  unsigned int u32x4;

#define B_   8
#define S_   1024
#define E_   128
#define H_   8
#define D_   16
#define F_   512
#define L_   3
#define MTOT (B_ * S_)   // 8192 rows of activations

// Async copy: 16B memory -> LDS (GV mode, 64-bit vaddr). INST_OFFSET is added
// to BOTH the LDS and memory addresses per the CDNA5 ISA pseudocode.
__device__ __forceinline__ void async_b128(unsigned lds_addr, const void* gptr) {
  unsigned long long ga = (unsigned long long)gptr;
  asm volatile("global_load_async_to_lds_b128 %0, %1, off"
               :: "v"(lds_addr), "v"(ga) : "memory");
}
__device__ __forceinline__ void async_b128_off16(unsigned lds_addr, const void* gptr) {
  unsigned long long ga = (unsigned long long)gptr;
  asm volatile("global_load_async_to_lds_b128 %0, %1, off offset:16"
               :: "v"(lds_addr), "v"(ga) : "memory");
}
__device__ __forceinline__ void wait_async0() {
  asm volatile("s_wait_asynccnt 0x0" ::: "memory");
}
__device__ __forceinline__ unsigned lds_off(const void* p) {
  return (unsigned)(uintptr_t)p;   // low 32 bits of generic addr = LDS offset
}

// ---------------------------------------------------------------------------
// fp32 -> bf16 weight conversion (grid-stride)
// ---------------------------------------------------------------------------
__global__ void convert_f32_to_bf16(const float* __restrict__ src,
                                    __bf16* __restrict__ dst, int n) {
  int i = blockIdx.x * blockDim.x + threadIdx.x;
  int stride = gridDim.x * blockDim.x;
  for (; i < n; i += stride) dst[i] = (__bf16)src[i];
}

// ---------------------------------------------------------------------------
// Tiled bf16 WMMA GEMM:  C = A @ Bw^T + bias (+ resid|normR(resid)) (+relu)
// ANORM : instance-norm fp32 A rows while staging (scale+shift).
// ABF16 : A already bf16 -> async-to-LDS staging (no VGPR round trip).
// OUTBF : store C as bf16.
// Block: 256 threads = 8 waves; block tile 128(M) x 64(N); K-step 32.
// ---------------------------------------------------------------------------
template<bool RELU, bool RES, bool RESNORM, bool ANORM, bool ABF16, bool OUTBF>
__global__ __launch_bounds__(256)
void gemm_bf16_kernel(const float* __restrict__ Af, const __bf16* __restrict__ Ab,
                      const __bf16* __restrict__ Bw,
                      const float* __restrict__ bias, const float* __restrict__ resid,
                      float* __restrict__ Cf, __bf16* __restrict__ Cb,
                      int M, int N, int K,
                      const float* __restrict__ nmean, const float* __restrict__ nrstd,
                      const float* __restrict__ nw, const float* __restrict__ nb)
{
  const int BK = 32;
  __shared__ __bf16 As[128][BK + 8];
  __shared__ __bf16 Bs[64][BK + 8];

  const int tid  = threadIdx.x;
  const int lane = tid & 31;
  const int wid  = tid >> 5;
  const int wm   = (wid & 3) * 32;
  const int wn   = (wid >> 2) * 32;
  const int m0   = blockIdx.y * 128;
  const int n0   = blockIdx.x * 64;
  const int bb   = m0 >> 10;          // batch index (1024-row batches)

  const int hl = lane >> 4;
  const int lr = lane & 15;
  const int k0 = hl * 8;

  v8f acc[2][2] = {};

  for (int kk = 0; kk < K; kk += BK) {
    // --- stage A tile 128x32 ---
    if (ABF16) {
      int r = tid >> 1;
      int c = (tid & 1) * 16;                       // 16 bf16 = 32B per thread
      unsigned l0 = lds_off(&As[r][c]);
      const __bf16* src = Ab + (size_t)(m0 + r) * K + kk + c;
      async_b128(l0, src);
      async_b128_off16(l0, src);
    } else {
      int r = tid >> 1;
      int c = (tid & 1) * 16;
      const f32x4* src4 = (const f32x4*)(Af + (size_t)(m0 + r) * K + kk + c);
      if (kk + BK < K) __builtin_prefetch((const float*)src4 + BK, 0, 0);
      f32x4 sc4[4], sh4[4];
      if (ANORM) {
        const f32x4* pm = (const f32x4*)(nmean + bb * E_ + kk + c);
        const f32x4* pr = (const f32x4*)(nrstd + bb * E_ + kk + c);
        const f32x4* pw = (const f32x4*)(nw + kk + c);
        const f32x4* pb = (const f32x4*)(nb + kk + c);
        #pragma unroll
        for (int j = 0; j < 4; ++j) {
          f32x4 sc = pr[j] * pw[j];
          sc4[j] = sc;
          sh4[j] = pb[j] - pm[j] * sc;
        }
      }
      #pragma unroll
      for (int j = 0; j < 4; ++j) {
        f32x4 v = src4[j];
        if (ANORM) v = v * sc4[j] + sh4[j];
        As[r][c + 4 * j + 0] = (__bf16)v.x;
        As[r][c + 4 * j + 1] = (__bf16)v.y;
        As[r][c + 4 * j + 2] = (__bf16)v.z;
        As[r][c + 4 * j + 3] = (__bf16)v.w;
      }
    }
    // --- stage B tile 64x32 (async 16B copies) ---
    {
      int r = tid >> 2;
      int c = (tid & 3) * 8;
      async_b128(lds_off(&Bs[r][c]), Bw + (size_t)(n0 + r) * K + kk + c);
    }
    wait_async0();
    __syncthreads();

    v16bf af[2], bfr[2];
    #pragma unroll
    for (int ti = 0; ti < 2; ++ti) {
      int row = wm + ti * 16 + lr;
      union { u32x4 u[2]; v16bf v; } t;
      t.u[0] = *(const u32x4*)&As[row][k0];
      t.u[1] = *(const u32x4*)&As[row][k0 + 16];
      af[ti] = t.v;
    }
    #pragma unroll
    for (int tj = 0; tj < 2; ++tj) {
      int row = wn + tj * 16 + lr;
      union { u32x4 u[2]; v16bf v; } t;
      t.u[0] = *(const u32x4*)&Bs[row][k0];
      t.u[1] = *(const u32x4*)&Bs[row][k0 + 16];
      bfr[tj] = t.v;
    }
    #pragma unroll
    for (int ti = 0; ti < 2; ++ti)
      #pragma unroll
      for (int tj = 0; tj < 2; ++tj)
        acc[ti][tj] = __builtin_amdgcn_wmma_f32_16x16x32_bf16(
            false, af[ti], false, bfr[tj], (short)0, acc[ti][tj], false, false);
    __syncthreads();
  }

  // --- epilogue: bias (+ (normalized) resid) (+relu), fp32 or bf16 store ---
  #pragma unroll
  for (int ti = 0; ti < 2; ++ti) {
    #pragma unroll
    for (int tj = 0; tj < 2; ++tj) {
      int col = n0 + wn + tj * 16 + lr;
      float bv = bias[col];
      float rm = 0.f, rr = 0.f, rwv = 0.f, rbv = 0.f;
      if (RESNORM) {
        rm  = nmean[bb * E_ + col];
        rr  = nrstd[bb * E_ + col];
        rwv = nw[col];
        rbv = nb[col];
      }
      #pragma unroll
      for (int r = 0; r < 8; ++r) {
        int row = m0 + wm + ti * 16 + r + hl * 8;
        float v = acc[ti][tj][r] + bv;
        if (RES) {
          float rv = resid[(size_t)row * N + col];
          if (RESNORM) rv = (rv - rm) * rr * rwv + rbv;
          v += rv;
        }
        if (RELU) v = fmaxf(v, 0.0f);
        if (OUTBF) Cb[(size_t)row * N + col] = (__bf16)v;
        else       Cf[(size_t)row * N + col] = v;
      }
    }
  }
}

// ---------------------------------------------------------------------------
// QKV GEMM (M=8192, N=384): fp32 A (+optional fused norm), bf16 split epilogue:
//   cols [0,256)  -> qk[row][col]        (bf16 row-major, Q then K)
//   cols [256,384)-> vT[col-256][row]    (bf16, head-dim-major for b128 reads)
// ---------------------------------------------------------------------------
template<bool ANORM>
__global__ __launch_bounds__(256)
void gemm_qkv_kernel(const float* __restrict__ A, const __bf16* __restrict__ Bw,
                     const float* __restrict__ bias,
                     __bf16* __restrict__ qk, __bf16* __restrict__ vT, int K,
                     const float* __restrict__ nmean, const float* __restrict__ nrstd,
                     const float* __restrict__ nw, const float* __restrict__ nb)
{
  const int BK = 32;
  __shared__ __bf16 As[128][BK + 8];
  __shared__ __bf16 Bs[64][BK + 8];

  const int tid  = threadIdx.x;
  const int lane = tid & 31;
  const int wid  = tid >> 5;
  const int wm   = (wid & 3) * 32;
  const int wn   = (wid >> 2) * 32;
  const int m0   = blockIdx.y * 128;
  const int n0   = blockIdx.x * 64;
  const int bb   = m0 >> 10;

  const int hl = lane >> 4;
  const int lr = lane & 15;
  const int k0 = hl * 8;

  v8f acc[2][2] = {};

  for (int kk = 0; kk < K; kk += BK) {
    {
      int r = tid >> 1;
      int c = (tid & 1) * 16;
      const f32x4* src4 = (const f32x4*)(A + (size_t)(m0 + r) * K + kk + c);
      if (kk + BK < K) __builtin_prefetch((const float*)src4 + BK, 0, 0);
      f32x4 sc4[4], sh4[4];
      if (ANORM) {
        const f32x4* pm = (const f32x4*)(nmean + bb * E_ + kk + c);
        const f32x4* pr = (const f32x4*)(nrstd + bb * E_ + kk + c);
        const f32x4* pw = (const f32x4*)(nw + kk + c);
        const f32x4* pb = (const f32x4*)(nb + kk + c);
        #pragma unroll
        for (int j = 0; j < 4; ++j) {
          f32x4 sc = pr[j] * pw[j];
          sc4[j] = sc;
          sh4[j] = pb[j] - pm[j] * sc;
        }
      }
      #pragma unroll
      for (int j = 0; j < 4; ++j) {
        f32x4 v = src4[j];
        if (ANORM) v = v * sc4[j] + sh4[j];
        As[r][c + 4 * j + 0] = (__bf16)v.x;
        As[r][c + 4 * j + 1] = (__bf16)v.y;
        As[r][c + 4 * j + 2] = (__bf16)v.z;
        As[r][c + 4 * j + 3] = (__bf16)v.w;
      }
    }
    {
      int r = tid >> 2;
      int c = (tid & 3) * 8;
      async_b128(lds_off(&Bs[r][c]), Bw + (size_t)(n0 + r) * K + kk + c);
    }
    wait_async0();
    __syncthreads();

    v16bf af[2], bfr[2];
    #pragma unroll
    for (int ti = 0; ti < 2; ++ti) {
      int row = wm + ti * 16 + lr;
      union { u32x4 u[2]; v16bf v; } t;
      t.u[0] = *(const u32x4*)&As[row][k0];
      t.u[1] = *(const u32x4*)&As[row][k0 + 16];
      af[ti] = t.v;
    }
    #pragma unroll
    for (int tj = 0; tj < 2; ++tj) {
      int row = wn + tj * 16 + lr;
      union { u32x4 u[2]; v16bf v; } t;
      t.u[0] = *(const u32x4*)&Bs[row][k0];
      t.u[1] = *(const u32x4*)&Bs[row][k0 + 16];
      bfr[tj] = t.v;
    }
    #pragma unroll
    for (int ti = 0; ti < 2; ++ti)
      #pragma unroll
      for (int tj = 0; tj < 2; ++tj)
        acc[ti][tj] = __builtin_amdgcn_wmma_f32_16x16x32_bf16(
            false, af[ti], false, bfr[tj], (short)0, acc[ti][tj], false, false);
    __syncthreads();
  }

  #pragma unroll
  for (int ti = 0; ti < 2; ++ti) {
    #pragma unroll
    for (int tj = 0; tj < 2; ++tj) {
      int col  = n0 + wn + tj * 16 + lr;
      float bv = bias[col];
      int row0 = m0 + wm + ti * 16 + hl * 8;   // rows row0..row0+7 in VGPRs 0..7
      if (col < 256) {
        #pragma unroll
        for (int r = 0; r < 8; ++r)
          qk[(size_t)(row0 + r) * 256 + col] = (__bf16)(acc[ti][tj][r] + bv);
      } else {
        union { __bf16 hh[8]; u32x4 u; } pk;
        #pragma unroll
        for (int r = 0; r < 8; ++r) pk.hh[r] = (__bf16)(acc[ti][tj][r] + bv);
        *(u32x4*)&vT[(size_t)(col - 256) * MTOT + row0] = pk.u;
      }
    }
  }
}

// ---------------------------------------------------------------------------
// Flash attention, transposed score layout; all operands bf16 / b128 loads.
// One wave per (b, h, 16 q rows); two 16-key sub-tiles fill K=32 of P*V.
// Output written as bf16 (consumed as the Wo GEMM's A operand).
// ---------------------------------------------------------------------------
__global__ __launch_bounds__(32)
void attn_kernel(const __bf16* __restrict__ qk, const __bf16* __restrict__ vT,
                 const float* __restrict__ adj, __bf16* __restrict__ out)
{
  __shared__ __bf16 pt[16 * 32];   // P[m][t] tile (m=query row, t=32 keys)

  const int lane = threadIdx.x;
  const int blk  = blockIdx.x;
  const int st   = blk & 63;
  const int h    = (blk >> 6) & 7;
  const int b    = blk >> 9;
  const int s0   = st * 16;

  const int hl = lane >> 4;
  const int lr = lane & 15;
  const int k0 = hl * 8;

  const u32x4 zu = {0u, 0u, 0u, 0u};

  union { u32x4 u[2]; v16bf v; } qu;
  qu.u[0] = *(const u32x4*)(qk + (size_t)(b * S_ + s0 + lr) * 256 + h * D_ + k0);
  qu.u[1] = zu;
  const v16bf qb = qu.v;

  v8f acc = {};
  float mrow = -3.0e38f;
  float lsum = 0.0f;

  for (int tt = 0; tt < 32; ++tt) {
    const int t0 = tt * 32;

    v8f c0 = {}, c1 = {};
    {
      union { u32x4 u[2]; v16bf v; } ku;
      ku.u[0] = *(const u32x4*)(qk + (size_t)(b * S_ + t0 + lr) * 256 + E_ + h * D_ + k0);
      ku.u[1] = zu;
      c0 = __builtin_amdgcn_wmma_f32_16x16x32_bf16(false, ku.v, false, qb, (short)0, c0, false, false);
    }
    {
      union { u32x4 u[2]; v16bf v; } ku;
      ku.u[0] = *(const u32x4*)(qk + (size_t)(b * S_ + t0 + 16 + lr) * 256 + E_ + h * D_ + k0);
      ku.u[1] = zu;
      c1 = __builtin_amdgcn_wmma_f32_16x16x32_bf16(false, ku.v, false, qb, (short)0, c1, false, false);
    }

    const float* abase = adj + ((size_t)(b * S_ + s0 + lr)) * S_ + t0 + 8 * hl;
    f32x4 a0 = *(const f32x4*)abase;
    f32x4 a1 = *(const f32x4*)(abase + 4);
    f32x4 a2 = *(const f32x4*)(abase + 16);
    f32x4 a3 = *(const f32x4*)(abase + 20);
    float av[16] = { a0.x, a0.y, a0.z, a0.w, a1.x, a1.y, a1.z, a1.w,
                     a2.x, a2.y, a2.z, a2.w, a3.x, a3.y, a3.z, a3.w };

    float s[16];
    #pragma unroll
    for (int r = 0; r < 8; ++r) {
      s[r]     = fminf(fmaxf(c0[r] * 0.25f * av[r],     -10000.0f), 10000.0f);
      s[8 + r] = fminf(fmaxf(c1[r] * 0.25f * av[8 + r], -10000.0f), 10000.0f);
    }

    float tmax = s[0];
    #pragma unroll
    for (int j = 1; j < 16; ++j) tmax = fmaxf(tmax, s[j]);
    tmax = fmaxf(tmax, __shfl_xor(tmax, 16, 32));

    float mnew  = fmaxf(mrow, tmax);
    float alpha = __expf(mrow - mnew);
    float p[16];
    float ps = 0.0f;
    #pragma unroll
    for (int j = 0; j < 16; ++j) { p[j] = __expf(s[j] - mnew); ps += p[j]; }
    ps += __shfl_xor(ps, 16, 32);
    lsum = lsum * alpha + ps;
    mrow = mnew;

    {
      union { __bf16 hh[8]; u32x4 u; } w0, w1;
      #pragma unroll
      for (int r = 0; r < 8; ++r) { w0.hh[r] = (__bf16)p[r]; w1.hh[r] = (__bf16)p[8 + r]; }
      *(u32x4*)&pt[lr * 32 + 8 * hl]      = w0.u;
      *(u32x4*)&pt[lr * 32 + 16 + 8 * hl] = w1.u;
    }

    #pragma unroll
    for (int r = 0; r < 8; ++r) acc[r] *= __shfl(alpha, r + 8 * hl, 32);

    union { u32x4 u[2]; v16bf v; } pau;
    pau.u[0] = *(const u32x4*)&pt[lr * 32 + k0];
    pau.u[1] = *(const u32x4*)&pt[lr * 32 + 16 + k0];

    const __bf16* vbase = vT + (size_t)(h * D_ + lr) * MTOT + b * S_ + t0;
    union { u32x4 u[2]; v16bf v; } vu;
    vu.u[0] = *(const u32x4*)(vbase + k0);
    vu.u[1] = *(const u32x4*)(vbase + 16 + k0);

    acc = __builtin_amdgcn_wmma_f32_16x16x32_bf16(false, pau.v, false, vu.v,
                                                  (short)0, acc, false, false);
  }

  #pragma unroll
  for (int r = 0; r < 8; ++r) {
    float lB = __shfl(lsum, r + 8 * hl, 32);
    int m = r + 8 * hl;
    out[((size_t)(b * S_ + s0 + m)) * E_ + h * D_ + lr] = (__bf16)(acc[r] / lB);
  }
}

// ---------------------------------------------------------------------------
// Instance norm statistics (biased variance), deterministic two-phase
// ---------------------------------------------------------------------------
__global__ __launch_bounds__(128)
void inorm_reduce(const float* __restrict__ y, float* __restrict__ psum,
                  float* __restrict__ psq)
{
  int b = blockIdx.x >> 3;
  int c = blockIdx.x & 7;
  int e = threadIdx.x;
  float s1 = 0.0f, s2 = 0.0f;
  for (int s = c * 128; s < c * 128 + 128; ++s) {
    float v = y[((size_t)(b * S_ + s)) * E_ + e];
    s1 += v;
    s2 += v * v;
  }
  psum[(b * 8 + c) * E_ + e] = s1;
  psq [(b * 8 + c) * E_ + e] = s2;
}

__global__ __launch_bounds__(128)
void inorm_stats(const float* __restrict__ psum, const float* __restrict__ psq,
                 float* __restrict__ mean, float* __restrict__ rstd)
{
  int b = blockIdx.x;
  int e = threadIdx.x;
  float s1 = 0.0f, s2 = 0.0f;
  #pragma unroll
  for (int c = 0; c < 8; ++c) {
    s1 += psum[(b * 8 + c) * E_ + e];
    s2 += psq [(b * 8 + c) * E_ + e];
  }
  float mu  = s1 * (1.0f / (float)S_);
  float var = s2 * (1.0f / (float)S_) - mu * mu;
  mean[b * E_ + e] = mu;
  rstd[b * E_ + e] = rsqrtf(var + 1e-5f);
}

__global__ __launch_bounds__(256)
void inorm_apply(const float* __restrict__ y, const float* __restrict__ mean,
                 const float* __restrict__ rstd, const float* __restrict__ w,
                 const float* __restrict__ bb, float* __restrict__ out)
{
  size_t idx = (size_t)blockIdx.x * blockDim.x + threadIdx.x;
  int e = (int)(idx & (E_ - 1));
  int b = (int)(idx >> 17);
  float v = y[idx];
  out[idx] = (v - mean[b * E_ + e]) * rstd[b * E_ + e] * w[e] + bb[e];
}

// ---------------------------------------------------------------------------
// Host launcher
// ---------------------------------------------------------------------------
extern "C" void kernel_launch(void* const* d_in, const int* in_sizes, int n_in,
                              void* d_out, int out_size, void* d_ws, size_t ws_size,
                              hipStream_t stream) {
  const float* x      = (const float*)d_in[0];
  const float* adj    = (const float*)d_in[1];
  const float* Wqkv_w = (const float*)d_in[2];
  const float* Wqkv_b = (const float*)d_in[3];
  const float* Wo_w   = (const float*)d_in[4];
  const float* Wo_b   = (const float*)d_in[5];
  const float* n1_w   = (const float*)d_in[6];
  const float* n1_b   = (const float*)d_in[7];
  const float* W1     = (const float*)d_in[8];
  const float* b1     = (const float*)d_in[9];
  const float* W2     = (const float*)d_in[10];
  const float* b2     = (const float*)d_in[11];
  const float* n2_w   = (const float*)d_in[12];
  const float* n2_b   = (const float*)d_in[13];
  float* outp = (float*)d_out;

  char* base = (char*)d_ws;
  size_t off = 0;
  auto take = [&](size_t bytes) { char* p = base + off; off += (bytes + 255) & ~(size_t)255; return p; };

  __bf16* wqkv_bf = (__bf16*)take((size_t)L_ * 384 * 128 * 2);
  __bf16* wo_bf   = (__bf16*)take((size_t)L_ * 128 * 128 * 2);
  __bf16* w1_bf   = (__bf16*)take((size_t)L_ * 512 * 128 * 2);
  __bf16* w2_bf   = (__bf16*)take((size_t)L_ * 128 * 512 * 2);
  __bf16* ffbuf   = (__bf16*)take((size_t)MTOT * F_ * 2);    // FFN hidden (bf16)
  __bf16* qkbuf   = (__bf16*)take((size_t)MTOT * 256 * 2);   // Q|K bf16 row-major
  __bf16* vbuf    = (__bf16*)take((size_t)E_ * MTOT * 2);    // V bf16, [h*d][B*S]
  __bf16* attnout = (__bf16*)take((size_t)MTOT * E_ * 2);    // attention out (bf16)
  float*  ybuf    = (float*)take((size_t)MTOT * E_ * 4);     // running activations
  float*  psum    = (float*)take((size_t)B_ * 8 * E_ * 4);
  float*  psq     = (float*)take((size_t)B_ * 8 * E_ * 4);
  float*  meanb   = (float*)take((size_t)B_ * E_ * 4);
  float*  rstdb   = (float*)take((size_t)B_ * E_ * 4);
  (void)ws_size; (void)n_in; (void)in_sizes; (void)out_size;

  convert_f32_to_bf16<<<256, 256, 0, stream>>>(Wqkv_w, wqkv_bf, L_ * 384 * 128);
  convert_f32_to_bf16<<<256, 256, 0, stream>>>(Wo_w,   wo_bf,   L_ * 128 * 128);
  convert_f32_to_bf16<<<256, 256, 0, stream>>>(W1,     w1_bf,   L_ * 512 * 128);
  convert_f32_to_bf16<<<256, 256, 0, stream>>>(W2,     w2_bf,   L_ * 128 * 512);

  const dim3 blk256(256), blk128(128), blk32(32);

  for (int i = 0; i < L_; ++i) {
    const bool first = (i == 0);
    // meanb/rstdb currently hold stats2 of layer i-1 (when i>0)

    // 1) qkv GEMM -> qkbuf (bf16) + vbuf (bf16 transposed); A normalized for i>0
    if (first)
      gemm_qkv_kernel<false><<<dim3(384 / 64, MTOT / 128), blk256, 0, stream>>>(
          x, wqkv_bf + (size_t)i * 384 * 128, Wqkv_b + i * 384,
          qkbuf, vbuf, 128, nullptr, nullptr, nullptr, nullptr);
    else
      gemm_qkv_kernel<true><<<dim3(384 / 64, MTOT / 128), blk256, 0, stream>>>(
          ybuf, wqkv_bf + (size_t)i * 384 * 128, Wqkv_b + i * 384,
          qkbuf, vbuf, 128, meanb, rstdb, n2_w + (i - 1) * E_, n2_b + (i - 1) * E_);

    // 2) attention -> attnout (bf16)
    attn_kernel<<<dim3(B_ * H_ * (S_ / 16)), blk32, 0, stream>>>(qkbuf, vbuf, adj, attnout);

    // 3) y1 = attnout @ Wo^T + bo + h_in   (h_in = x, or normalized prev ybuf)
    if (first)
      gemm_bf16_kernel<false, true, false, false, true, false>
          <<<dim3(128 / 64, MTOT / 128), blk256, 0, stream>>>(
          nullptr, attnout, wo_bf + (size_t)i * 128 * 128, Wo_b + i * 128, x,
          ybuf, nullptr, MTOT, 128, 128, nullptr, nullptr, nullptr, nullptr);
    else
      gemm_bf16_kernel<false, true, true, false, true, false>
          <<<dim3(128 / 64, MTOT / 128), blk256, 0, stream>>>(
          nullptr, attnout, wo_bf + (size_t)i * 128 * 128, Wo_b + i * 128, ybuf,
          ybuf, nullptr, MTOT, 128, 128, meanb, rstdb, n2_w + (i - 1) * E_, n2_b + (i - 1) * E_);

    // 4) stats1 of y1
    inorm_reduce<<<dim3(B_ * 8), blk128, 0, stream>>>(ybuf, psum, psq);
    inorm_stats <<<dim3(B_),     blk128, 0, stream>>>(psum, psq, meanb, rstdb);

    // 5) ff = relu(norm1(y1) @ W1^T + b1)  -> bf16 (A normalized while staging)
    gemm_bf16_kernel<true, false, false, true, false, true>
        <<<dim3(512 / 64, MTOT / 128), blk256, 0, stream>>>(
        ybuf, nullptr, w1_bf + (size_t)i * 512 * 128, b1 + i * 512, nullptr,
        nullptr, ffbuf, MTOT, 512, 128, meanb, rstdb, n1_w + i * E_, n1_b + i * E_);

    // 6) y2 = ff @ W2^T + b2 + norm1(y1)   (bf16 A, residual normalized)
    gemm_bf16_kernel<false, true, true, false, true, false>
        <<<dim3(128 / 64, MTOT / 128), blk256, 0, stream>>>(
        nullptr, ffbuf, w2_bf + (size_t)i * 128 * 512, b2 + i * 128, ybuf,
        ybuf, nullptr, MTOT, 128, 512, meanb, rstdb, n1_w + i * E_, n1_b + i * E_);

    // 7) stats2 of y2 (consumed by next layer, or by the final apply)
    inorm_reduce<<<dim3(B_ * 8), blk128, 0, stream>>>(ybuf, psum, psq);
    inorm_stats <<<dim3(B_),     blk128, 0, stream>>>(psum, psq, meanb, rstdb);
  }

  // final norm2 materialized to d_out
  inorm_apply<<<dim3(MTOT * E_ / 256), blk256, 0, stream>>>(
      ybuf, meanb, rstdb, n2_w + (L_ - 1) * E_, n2_b + (L_ - 1) * E_, outp);
}
